// Decoder_446676599461
// MI455X (gfx1250) — matrix operand order repaired and checked by reading, not compiled, not run on hardware
//
#include <hip/hip_runtime.h>
#include <hip/hip_bf16.h>
#include <cmath>

typedef __bf16 bf16_t;
typedef __attribute__((ext_vector_type(16))) __bf16 v16bf;
typedef __attribute__((ext_vector_type(8)))  float  v8f;

#define Bb 64
#define Ss 128
#define Tt 48
#define Ee 512
#define Hh 512
#define EMBd 512
#define VOCABn 32000
#define SH 16               // batch rows per scan workgroup
#define SCAN_THREADS 512
#define SCAN_WAVES (SCAN_THREADS / 32)

// ---------------- WMMA helpers (bf16 16x16x32, f32 accumulate) ----------------

__device__ __forceinline__ v8f wmma_bf16(v16bf a, v16bf b, v8f c) {
  return __builtin_amdgcn_wmma_f32_16x16x32_bf16(false, a, false, b, (short)0, c,
                                                 false, false);
}

// A fragment (16xK tile, row M = lane&15). lane<16: K={kk..kk+7, kk+16..kk+23};
// lane>=16: K={kk+8..kk+15, kk+24..kk+31}.
__device__ __forceinline__ v16bf load_a_bf16(const bf16_t* src, int ld, int kk, int lane) {
  int half = lane >> 4, lrow = lane & 15;
  const bf16_t* r = src + lrow * ld;
  int k0 = kk + half * 8;
  int k1 = kk + 16 + half * 8;
  v16bf a;
#pragma unroll
  for (int i = 0; i < 8; ++i) a[i] = r[k0 + i];
#pragma unroll
  for (int i = 0; i < 8; ++i) a[8 + i] = r[k1 + i];
  return a;
}

// A fragment from an LDS-staged f32 tile with leading dim 32 (one K-chunk).
__device__ __forceinline__ v16bf load_a_lds_f32(const float* src, int r0, int lane) {
  int half = lane >> 4, lrow = lane & 15;
  const float* r = src + (r0 + lrow) * 32;
  v16bf a;
#pragma unroll
  for (int i = 0; i < 8; ++i) a[i] = (bf16_t)r[half * 8 + i];
#pragma unroll
  for (int i = 0; i < 8; ++i) a[8 + i] = (bf16_t)r[16 + half * 8 + i];
  return a;
}

// B fragment for D = A * W^T (W row-major [n][k], K contiguous).
// lane<16: col n=n0+lane, K={kk..kk+7, kk+16..kk+23} ; lane>=16: col n=n0+lane-16, K={kk+8..kk+15, kk+24..kk+31}.
__device__ __forceinline__ v16bf load_bT_f32(const float* w, int ld, int n0, int kk, int lane) {
  int half = lane >> 4, lrow = lane & 15;
  const float* r = w + (size_t)(n0 + lrow) * ld + kk + half * 8;
  v16bf b;
#pragma unroll
  for (int i = 0; i < 16; ++i) b[i] = (bf16_t)r[(i & 7) + (i >> 3) * 16];
  return b;
}

__device__ __forceinline__ v16bf load_bT_bf16(const bf16_t* w, int ld, int n0, int kk, int lane) {
  int half = lane >> 4, lrow = lane & 15;
  const bf16_t* r = w + (size_t)(n0 + lrow) * ld + kk + half * 8;
  v16bf b;
#pragma unroll
  for (int i = 0; i < 16; ++i) b[i] = r[(i & 7) + (i >> 3) * 16];
  return b;
}

// Async global->LDS copy of 16 bytes (ASYNCcnt-tracked LDS DMA).
__device__ __forceinline__ void async_ld_b128(unsigned lds_addr, const float* g) {
  asm volatile("global_load_async_to_lds_b128 %0, %1, off"
               :: "v"(lds_addr), "v"(g) : "memory");
}

// ---------------- kernel 1: h0 = [enc_hidden | style[:, -1]] @ Wp.T + bp -----

__global__ __launch_bounds__(512) void k_h0(const float* __restrict__ enc_h,
                                            const float* __restrict__ style,
                                            const float* __restrict__ Wp,
                                            const float* __restrict__ bp,
                                            float* __restrict__ h0) {
  int b = blockIdx.x;
  int j = threadIdx.x;
  const float* w = Wp + (size_t)j * (Ee + 128);
  const float* eh = enc_h + (size_t)b * Ee;
  const float* sv = style + ((size_t)b * 4 + 3) * 128;
  float acc = bp[j];
  for (int k = 0; k < Ee; ++k) acc += eh[k] * w[k];
  for (int k = 0; k < 128; ++k) acc += sv[k] * w[Ee + k];
  h0[(size_t)b * Hh + j] = acc;
}

// ---------------- kernel 2: attn_w (K-major strided) -> bf16 transpose -------

__global__ __launch_bounds__(256) void k_tr(const float* __restrict__ aw,
                                            bf16_t* __restrict__ awT) {
  int i = blockIdx.x * 256 + threadIdx.x;   // over 512*512
  int k = i >> 9, e = i & 511;              // aw[k][e]
  awT[(size_t)e * Hh + k] = (bf16_t)aw[i];
}

// ---------------- kernel 3: fused attention + GRU scan (48 steps) ------------

__global__ __launch_bounds__(SCAN_THREADS) void k_scan(
    const float* __restrict__ enc,           // (B,S,E)
    const unsigned char* __restrict__ mask,  // (B,S) bool
    const int* __restrict__ dec_in,          // (B,T)
    const float* __restrict__ emb,           // (VOCAB, EMB)
    const bf16_t* __restrict__ awT,          // (E,H) bf16 = attn_w^T
    const float* __restrict__ W_ih,          // (3H, E+EMB)
    const float* __restrict__ W_hh,          // (3H, H)
    const float* __restrict__ b_ih, const float* __restrict__ b_hh,
    const float* __restrict__ h0,            // (B,H)
    float* __restrict__ hs)                  // (B,T,H)
{
  // 112 KB LDS pool, phase-disjoint buffers aliased:
  //   [0,16K)   sh_hb   h (bf16, 16x512)
  //   [16,48K)  sh_x    [ctx | emb] (bf16, 16x1024)
  //   [48,80K)  sh_rz   sigmoid gates r,z (bf16, 16x1024)
  //   [80,96K)  sh_q (bf16 16x512)  ALIAS  sh_gin (bf16 16x512)
  //   [96,112K) sh_ghn (bf16 16x512) ALIAS sh_s (f32 16x128, first 8K)
  __shared__ __align__(16) char smem[112 * 1024];
  bf16_t* sh_hb  = (bf16_t*)(smem);
  bf16_t* sh_x   = (bf16_t*)(smem + 16 * 1024);
  bf16_t* sh_rz  = (bf16_t*)(smem + 48 * 1024);
  bf16_t* sh_q   = (bf16_t*)(smem + 80 * 1024);
  bf16_t* sh_gin = (bf16_t*)(smem + 80 * 1024);
  bf16_t* sh_ghn = (bf16_t*)(smem + 96 * 1024);
  float*  sh_s   = (float*) (smem + 96 * 1024);

  const int tid  = threadIdx.x;
  const int lane = tid & 31;
  const int wave = tid >> 5;
  const int b0   = blockIdx.x * SH;

  for (int i = tid; i < SH * Hh; i += SCAN_THREADS) {
    int b = i >> 9, j = i & 511;
    sh_hb[i] = (bf16_t)h0[(size_t)(b0 + b) * Hh + j];
  }
  __syncthreads();

  for (int t = 0; t < Tt; ++t) {
    // ---- q = h @ attn_w : M=16, N=512, K=512 (WMMA) ----
    for (int nt = wave; nt < Ee / 16; nt += SCAN_WAVES) {
      int n0 = nt * 16;
      v8f acc = {};
      for (int kk = 0; kk < Hh; kk += 32) {
        v16bf a = load_a_bf16(sh_hb, Hh, kk, lane);
        v16bf b = load_bT_bf16(awT, Hh, n0, kk, lane);
        acc = wmma_bf16(a, b, acc);
      }
      int nl = n0 + (lane & 15);
      int mh = (lane >> 4) * 8;
#pragma unroll
      for (int j = 0; j < 8; ++j) sh_q[(mh + j) * Ee + nl] = (bf16_t)acc[j];
    }
    __syncthreads();

    // ---- scores[b,s] = q[b,:] . enc[b,s,:] with mask ----
    {
      int b = tid >> 5;          // 0..15
      int s0 = tid & 31;
      const bf16_t* qb = sh_q + b * Ee;
      for (int s = s0; s < Ss; s += 32) {
        const float* e = enc + (((size_t)(b0 + b)) * Ss + s) * Ee;
        float acc = 0.f;
        for (int k = 0; k < Ee; ++k) acc += (float)qb[k] * e[k];
        sh_s[b * Ss + s] = mask[(size_t)(b0 + b) * Ss + s] ? -INFINITY : acc;
      }
    }
    __syncthreads();

    // ---- softmax over S (one thread per batch row; S=128 serial) ----
    if (tid < SH) {
      float* row = sh_s + tid * Ss;
      float m = -INFINITY;
      for (int s = 0; s < Ss; ++s) m = fmaxf(m, row[s]);
      float sum = 0.f;
      for (int s = 0; s < Ss; ++s) { float v = __expf(row[s] - m); row[s] = v; sum += v; }
      float inv = 1.f / sum;
      for (int s = 0; s < Ss; ++s) row[s] *= inv;
    }
    __syncthreads();

    // ---- ctx + embedding lookup -> x = [ctx | emb] (bf16) ----
    for (int i = tid; i < SH * Ee; i += SCAN_THREADS) {
      int b = i >> 9, e0 = i & 511;
      const float* eb = enc + ((size_t)(b0 + b) * Ss) * Ee + e0;
      const float* pb = sh_s + b * Ss;
      float acc = 0.f;
      for (int s = 0; s < Ss; ++s) acc += pb[s] * eb[(size_t)s * Ee];
      sh_x[b * (Ee + EMBd) + e0] = (bf16_t)acc;
    }
    for (int i = tid; i < SH * EMBd; i += SCAN_THREADS) {
      int b = i >> 9, e0 = i & 511;
      int tok = dec_in[(size_t)(b0 + b) * Tt + t];
      sh_x[b * (Ee + EMBd) + Ee + e0] = (bf16_t)emb[(size_t)tok * EMBd + e0];
    }
    __syncthreads();

    // ---- GRU gates: gi = x @ W_ih^T (K=1024), gh = h @ W_hh^T (K=512) ----
    for (int nt = wave; nt < (3 * Hh) / 16; nt += SCAN_WAVES) {
      int n0 = nt * 16;
      v8f gi = {};
      for (int kk = 0; kk < Ee + EMBd; kk += 32) {
        v16bf a = load_a_bf16(sh_x, Ee + EMBd, kk, lane);
        v16bf b = load_bT_f32(W_ih, Ee + EMBd, n0, kk, lane);
        gi = wmma_bf16(a, b, gi);
      }
      v8f gh = {};
      for (int kk = 0; kk < Hh; kk += 32) {
        v16bf a = load_a_bf16(sh_hb, Hh, kk, lane);
        v16bf b = load_bT_f32(W_hh, Hh, n0, kk, lane);
        gh = wmma_bf16(a, b, gh);
      }
      int nl = n0 + (lane & 15);
      int mh = (lane >> 4) * 8;
      if (n0 < 2 * Hh) {  // r,z region: sigmoid(gi+gh+biases)
        float bias = b_ih[nl] + b_hh[nl];
#pragma unroll
        for (int j = 0; j < 8; ++j) {
          float g = gi[j] + gh[j] + bias;
          sh_rz[(mh + j) * (2 * Hh) + nl] = (bf16_t)(1.f / (1.f + __expf(-g)));
        }
      } else {            // n region: keep gi_n, gh_n separate
        float bi = b_ih[nl], bh = b_hh[nl];
        int nn = nl - 2 * Hh;
#pragma unroll
        for (int j = 0; j < 8; ++j) {
          sh_gin[(mh + j) * Hh + nn] = (bf16_t)(gi[j] + bi);
          sh_ghn[(mh + j) * Hh + nn] = (bf16_t)(gh[j] + bh);
        }
      }
    }
    __syncthreads();

    // ---- h update: n = tanh(gin + r*ghn); h = (1-z)*n + z*h ----
    for (int i = tid; i < SH * Hh; i += SCAN_THREADS) {
      int b = i >> 9, j = i & 511;
      float r = (float)sh_rz[b * (2 * Hh) + j];
      float z = (float)sh_rz[b * (2 * Hh) + Hh + j];
      float nv = (float)sh_gin[i] + r * (float)sh_ghn[i];
      float e2 = __expf(-2.f * nv);
      float tn = (1.f - e2) / (1.f + e2);
      float hn = (1.f - z) * tn + z * (float)sh_hb[i];
      hs[(((size_t)(b0 + b)) * Tt + t) * Hh + j] = hn;
      sh_hb[i] = (bf16_t)hn;
    }
    __syncthreads();
  }
}

// ---------------- kernel 4: logits = hs @ Wo^T + bo (3072 x 32000 x 512) -----
// Block = 8 waves, covers M=64 x N=128. The shared hs A-tile (64 rows x 32 K,
// 8 KB) is staged into LDS once per block via async LDS-DMA, double-buffered;
// each wave's 4 M-tiles share one Wo B fragment.

__global__ __launch_bounds__(256) void k_logits(const float* __restrict__ hs,
                                                const float* __restrict__ Wo,
                                                const float* __restrict__ bo,
                                                float* __restrict__ out) {
  __shared__ __align__(16) float sh_a[2][64 * 32];   // 2 x 8 KB staging buffers
  const int tid  = threadIdx.x;
  const int lane = tid & 31;
  const int wave = tid >> 5;
  const int n0 = blockIdx.x * 128 + wave * 16;
  const int m0 = blockIdx.y * 64;

  // Stage K-chunk kt (cols kt*32..kt*32+31 of rows m0..m0+63) into sh_a[kt&1].
  // 512 x 16B chunks over 256 threads: 2 async b128 per thread, in order.
  auto stage = [&](int kt) {
    const float* src = hs + (size_t)m0 * Hh + kt * 32;
    unsigned base = (unsigned)(size_t)(&sh_a[kt & 1][0]);
    for (int c = tid; c < 512; c += 256) {
      int row = c >> 3, o16 = c & 7;            // 8 x 16B per 128B row
      async_ld_b128(base + (unsigned)c * 16u, src + (size_t)row * Hh + o16 * 4);
    }
  };

  v8f acc0 = {}, acc1 = {}, acc2 = {}, acc3 = {};
  stage(0);
  for (int kt = 0; kt < Hh / 32; ++kt) {
    if (kt < Hh / 32 - 1) {
      stage(kt + 1);                                   // prefetch next buffer
      asm volatile("s_wait_asynccnt 0x2" ::: "memory"); // oldest 2 (cur) done
    } else {
      asm volatile("s_wait_asynccnt 0x0" ::: "memory");
    }
    __syncthreads();                                   // publish current buffer

    const float* abuf = &sh_a[kt & 1][0];
    v16bf bf = load_bT_f32(Wo, Hh, n0, kt * 32, lane);
    v16bf a0 = load_a_lds_f32(abuf, 0,  lane);
    v16bf a1 = load_a_lds_f32(abuf, 16, lane);
    v16bf a2 = load_a_lds_f32(abuf, 32, lane);
    v16bf a3 = load_a_lds_f32(abuf, 48, lane);
    acc0 = wmma_bf16(a0, bf, acc0);
    acc1 = wmma_bf16(a1, bf, acc1);
    acc2 = wmma_bf16(a2, bf, acc2);
    acc3 = wmma_bf16(a3, bf, acc3);
    __syncthreads();                                   // done reading cur buffer
  }

  int nl = n0 + (lane & 15);
  int mh = (lane >> 4) * 8;
  float bias = bo[nl];
#pragma unroll
  for (int j = 0; j < 8; ++j) {
    out[(size_t)(m0 + mh + j) * VOCABn + nl]      = acc0[j] + bias;
    out[(size_t)(m0 + 16 + mh + j) * VOCABn + nl] = acc1[j] + bias;
    out[(size_t)(m0 + 32 + mh + j) * VOCABn + nl] = acc2[j] + bias;
    out[(size_t)(m0 + 48 + mh + j) * VOCABn + nl] = acc3[j] + bias;
  }
}

// ---------------- host ----------------

extern "C" void kernel_launch(void* const* d_in, const int* in_sizes, int n_in,
                              void* d_out, int out_size, void* d_ws, size_t ws_size,
                              hipStream_t stream) {
  (void)in_sizes; (void)n_in; (void)out_size; (void)ws_size;
  const float* enc_h          = (const float*)d_in[0];
  const float* enc            = (const float*)d_in[1];
  const unsigned char* emask  = (const unsigned char*)d_in[2];
  // d_in[3] target_seq: unused by forward
  const int*   dec_in         = (const int*)d_in[4];
  const float* style          = (const float*)d_in[5];
  const float* emb            = (const float*)d_in[6];
  const float* Wp             = (const float*)d_in[7];
  const float* bp             = (const float*)d_in[8];
  const float* attn_w         = (const float*)d_in[9];
  const float* W_ih           = (const float*)d_in[10];
  const float* W_hh           = (const float*)d_in[11];
  const float* b_ih           = (const float*)d_in[12];
  const float* b_hh           = (const float*)d_in[13];
  const float* Wo             = (const float*)d_in[14];
  const float* bo             = (const float*)d_in[15];
  float* out = (float*)d_out;

  char* ws = (char*)d_ws;
  float*  hs  = (float*)ws;                                          // 64*48*512 f32
  float*  h0  = (float*)(ws + (size_t)Bb * Tt * Hh * sizeof(float)); // 64*512 f32
  bf16_t* awT = (bf16_t*)(ws + ((size_t)Bb * Tt * Hh + (size_t)Bb * Hh) * sizeof(float));

  k_h0<<<Bb, 512, 0, stream>>>(enc_h, style, Wp, bp, h0);
  k_tr<<<(Hh * Ee) / 256, 256, 0, stream>>>(attn_w, awT);
  k_scan<<<Bb / SH, SCAN_THREADS, 0, stream>>>(enc, emask, dec_in, emb, awT,
                                               W_ih, W_hh, b_ih, b_hh, h0, hs);
  k_logits<<<dim3(VOCABn / 128, (Bb * Tt) / 64), 256, 0, stream>>>(hs, Wo, bo, out);
}
